// ScaledDotProductAttention_67860483277033
// MI455X (gfx1250) — matrix operand order repaired and checked by reading, not compiled, not run on hardware
//
#include <hip/hip_runtime.h>

typedef __attribute__((ext_vector_type(16))) _Float16 v16h;
typedef __attribute__((ext_vector_type(8)))  float    v8f;

#define B_    4
#define H_    16
#define S_    2048
#define D_    64
#define NT_   (S_ / 64)
#define SCALE 0.125f   // 1/sqrt(64)

// 16 half values (8 VGPRs) with 32-bit lane-pair access for packing
union Frag16 { v16h h; unsigned int u[8]; };
union Pack2  { _Float16 f[2]; unsigned int u; };

// DPP8 selector: per-lane source within each 8-lane group
#define DPP8_SEL(a,b,c,d,e,f,g,h) \
    ((a)|((b)<<3)|((c)<<6)|((d)<<9)|((e)<<12)|((f)<<15)|((g)<<18)|((h)<<21))
#define DPP8_XOR1 DPP8_SEL(1,0,3,2,5,4,7,6)
#define DPP8_XOR2 DPP8_SEL(2,3,0,1,6,7,4,5)
#define DPP8_XOR4 DPP8_SEL(4,5,6,7,0,1,2,3)
#define DPP16_ROW_ROR8 0x128   // rotate 16-lane row by 8 == xor 8 within the row

// Butterfly reductions over each 16-lane half of the wave32 — pure VALU (DPP),
// no LDS pipe, no dscnt waits (vs ds_bpermute from __shfl_xor).
__device__ __forceinline__ float red16_max(float x) {
    x = fmaxf(x, __int_as_float(__builtin_amdgcn_mov_dpp8(__float_as_int(x), DPP8_XOR1)));
    x = fmaxf(x, __int_as_float(__builtin_amdgcn_mov_dpp8(__float_as_int(x), DPP8_XOR2)));
    x = fmaxf(x, __int_as_float(__builtin_amdgcn_mov_dpp8(__float_as_int(x), DPP8_XOR4)));
    x = fmaxf(x, __int_as_float(__builtin_amdgcn_mov_dpp(
                     __float_as_int(x), DPP16_ROW_ROR8, 0xf, 0xf, true)));
    return x;
}
__device__ __forceinline__ float red16_sum(float x) {
    x += __int_as_float(__builtin_amdgcn_mov_dpp8(__float_as_int(x), DPP8_XOR1));
    x += __int_as_float(__builtin_amdgcn_mov_dpp8(__float_as_int(x), DPP8_XOR2));
    x += __int_as_float(__builtin_amdgcn_mov_dpp8(__float_as_int(x), DPP8_XOR4));
    x += __int_as_float(__builtin_amdgcn_mov_dpp(
             __float_as_int(x), DPP16_ROW_ROR8, 0xf, 0xf, true));
    return x;
}

// Issue this lane's share (2 x B128) of one wave's 16x64-byte mask tile:
// async DMA global -> LDS, tracked with ASYNCcnt (no VGPR round trip).
__device__ __forceinline__ void issue_mask_async(const unsigned char* grow,
                                                 unsigned ldsrow, int half) {
    const unsigned char* g0 = grow + half * 16;        // segment `half`
    const unsigned char* g1 = grow + half * 16 + 32;   // segment `half+2`
    unsigned l0 = ldsrow + half * 16;
    unsigned l1 = l0 + 32;
    asm volatile("global_load_async_to_lds_b128 %0, %1, off"
                 :: "v"(l0), "v"(g0) : "memory");
    asm volatile("global_load_async_to_lds_b128 %0, %1, off"
                 :: "v"(l1), "v"(g1) : "memory");
}

__global__ __launch_bounds__(128)
void flash_attn_f16wmma(const float* __restrict__ Q,
                        const float* __restrict__ K,
                        const float* __restrict__ V,
                        const unsigned char* __restrict__ Mask,
                        float* __restrict__ O)
{
    // LDS: K (row-major f16), V (d-major f16), double-buffered per-wave mask
    // tiles (async-DMA target), per-wave P staging for C->A layout conversion.
    __shared__ _Float16 Ks[64 * 72];                       // [krow][feat]
    __shared__ _Float16 Vs[64 * 72];                       // [d][krow]
    __shared__ __align__(16) unsigned char Msw[2 * 4 * 16 * 64];
    __shared__ _Float16 Ps[4 * 16 * 72];                   // per-wave [m][k]

    const int tid  = threadIdx.x;
    const int wave = tid >> 5;
    const int lane = tid & 31;
    const int half = lane >> 4;   // wave32: two 16-lane halves
    const int n16  = lane & 15;

    const int bh    = blockIdx.y;
    const int qbase = blockIdx.x * 64;

    const float* Qb = Q + (size_t)bh * S_ * D_;
    const float* Kb = K + (size_t)bh * S_ * D_;
    const float* Vb = V + (size_t)bh * S_ * D_;
    const unsigned char* Mb = Mask + (size_t)bh * S_ * S_;
    float* Ob = O + (size_t)bh * S_ * D_;

    // Per-lane mask source row (this wave's row lr = lane&15) and per-wave LDS bases.
    const unsigned char* mrow_g =
        Mb + (size_t)(qbase + wave * 16 + n16) * S_;       // + kt*64 per tile
    const unsigned mlds0 =
        (unsigned)(unsigned long long)(&Msw[0]) + wave * 1024 + n16 * 64;
    const unsigned char* Mr = &Msw[0] + wave * 1024;       // + (kt&1)*4096

    // ---- Load Q fragment once (A-layout, 16x64 -> two 16x32 f16 fragments) ----
    Frag16 aQ[2];
    {
        const int qrow = qbase + wave * 16 + n16;   // A rows: M = lane&15 (both halves)
        const float* qr = Qb + (size_t)qrow * D_;
#pragma unroll
        for (int c = 0; c < 2; ++c) {
#pragma unroll
            for (int v = 0; v < 8; ++v) {
                // A 16x32 layout: v0-3 hold K 0-7 (lanes 0-15) / 8-15 (lanes 16-31),
                //                 v4-7 hold K 16-23 / 24-31
                const int k = 32 * c + ((v >> 2) << 4) + (half << 3) + ((v & 3) << 1);
                float2 q2 = *(const float2*)(qr + k);
                Pack2 p; p.f[0] = (_Float16)q2.x; p.f[1] = (_Float16)q2.y;
                aQ[c].u[v] = p.u;
            }
        }
    }

    // ---- Flash state ----
    const v8f vzero = {0.f, 0.f, 0.f, 0.f, 0.f, 0.f, 0.f, 0.f};
    v8f acc[4];
#pragma unroll
    for (int t = 0; t < 4; ++t) acc[t] = vzero;
    float mrow[8], lrow[8];
#pragma unroll
    for (int r = 0; r < 8; ++r) { mrow[r] = -3.0e38f; lrow[r] = 0.f; }

    _Float16* myP = Ps + wave * 16 * 72;

    // Prologue: async-prefetch mask tile 0 into buffer 0.
    issue_mask_async(mrow_g, mlds0, half);

    for (int kt = 0; kt < NT_; ++kt) {
        // ---- Async-prefetch next mask tile into the other buffer ----
        if (kt + 1 < NT_)
            issue_mask_async(mrow_g + (kt + 1) * 64,
                             mlds0 + ((kt + 1) & 1) * 4096, half);

        // ---- Cooperative loads: K tile, V tile (transposed) ----
        {
            const float* Kt = Kb + (size_t)kt * 64 * D_;
            const float* Vt = Vb + (size_t)kt * 64 * D_;
#pragma unroll
            for (int i = 0; i < 8; ++i) {
                const int idx = tid + i * 128;      // 0..1023
                const int row = idx >> 4;           // 0..63
                const int fg  = (idx & 15) << 2;    // feature group * 4
                float4 k4 = *(const float4*)(Kt + row * D_ + fg);
                float4 v4 = *(const float4*)(Vt + row * D_ + fg);
                Ks[row * 72 + fg + 0] = (_Float16)k4.x;
                Ks[row * 72 + fg + 1] = (_Float16)k4.y;
                Ks[row * 72 + fg + 2] = (_Float16)k4.z;
                Ks[row * 72 + fg + 3] = (_Float16)k4.w;
                Vs[(fg + 0) * 72 + row] = (_Float16)v4.x;
                Vs[(fg + 1) * 72 + row] = (_Float16)v4.y;
                Vs[(fg + 2) * 72 + row] = (_Float16)v4.z;
                Vs[(fg + 3) * 72 + row] = (_Float16)v4.w;
            }
        }
        __syncthreads();

        // ---- S = Q * K^T  (16 x 64 tile per wave, f16 WMMA, f32 accum) ----
        v8f s[4];
#pragma unroll
        for (int t = 0; t < 4; ++t) {
            Frag16 b0, b1;
#pragma unroll
            for (int v = 0; v < 8; ++v) {
                // B 32x16 layout: v0-3 hold K 0-7 (lanes 0-15) / 8-15 (lanes 16-31), v4-7 hold K 16-23 / 24-31
                const int k0 = ((v >> 2) << 4) + (half << 3) + ((v & 3) << 1);
                b0.u[v] = *(const unsigned int*)&Ks[(t * 16 + n16) * 72 + k0];
                b1.u[v] = *(const unsigned int*)&Ks[(t * 16 + n16) * 72 + 32 + k0];
            }
            v8f c0 = __builtin_amdgcn_wmma_f32_16x16x32_f16(
                false, aQ[0].h, false, b0.h, (short)0, vzero, false, false);
            s[t] = __builtin_amdgcn_wmma_f32_16x16x32_f16(
                false, aQ[1].h, false, b1.h, (short)0, c0, false, false);
        }

        // ---- Drain async DMA: current mask tile resident (next may be in flight) ----
        if (kt + 1 < NT_) asm volatile("s_wait_asynccnt 2" ::: "memory");
        else              asm volatile("s_wait_asynccnt 0" ::: "memory");
        const unsigned char* Mt = Mr + (kt & 1) * 4096;

        // ---- Online softmax (rows r and r+8 per lane-half), DPP reductions ----
#pragma unroll
        for (int r = 0; r < 8; ++r) {
            const int lr = r + 8 * half;              // row within wave's 16-row tile
            float xm = -3.0e38f;
#pragma unroll
            for (int t = 0; t < 4; ++t) {
                float x = s[t][r] * SCALE;
                if (Mt[lr * 64 + t * 16 + n16]) x = -1.0e9f;  // mask==True -> -1e9
                s[t][r] = x;
                xm = fmaxf(xm, x);
            }
            xm = red16_max(xm);
            const float mn    = fmaxf(mrow[r], xm);
            const float alpha = __expf(mrow[r] - mn);
            mrow[r] = mn;
            float rs = 0.f;
#pragma unroll
            for (int t = 0; t < 4; ++t) {
                const float p = __expf(s[t][r] - mn);
                s[t][r] = p;
                rs += p;
            }
            rs = red16_sum(rs);
            lrow[r] = lrow[r] * alpha + rs;
#pragma unroll
            for (int t = 0; t < 4; ++t) acc[t][r] *= alpha;
        }

        // ---- C-layout P (f32) -> per-wave LDS -> A-layout f16 fragment ----
#pragma unroll
        for (int t = 0; t < 4; ++t)
#pragma unroll
            for (int r = 0; r < 8; ++r)
                myP[(r + 8 * half) * 72 + t * 16 + n16] = (_Float16)s[t][r];
        asm volatile("s_wait_dscnt 0" ::: "memory");

        Frag16 aP[2];
#pragma unroll
        for (int c = 0; c < 2; ++c)
#pragma unroll
            for (int v = 0; v < 8; ++v) {
                const int k = 32 * c + ((v >> 2) << 4) + (half << 3) + ((v & 3) << 1);
                aP[c].u[v] = *(const unsigned int*)&myP[n16 * 72 + k];
            }

        // ---- O += P * V  (V stored transposed: pairs along K contiguous) ----
#pragma unroll
        for (int t = 0; t < 4; ++t) {
            Frag16 b0, b1;
#pragma unroll
            for (int v = 0; v < 8; ++v) {
                const int k0 = ((v >> 2) << 4) + (half << 3) + ((v & 3) << 1);
                b0.u[v] = *(const unsigned int*)&Vs[(t * 16 + n16) * 72 + k0];
                b1.u[v] = *(const unsigned int*)&Vs[(t * 16 + n16) * 72 + 32 + k0];
            }
            v8f c0 = __builtin_amdgcn_wmma_f32_16x16x32_f16(
                false, aP[0].h, false, b0.h, (short)0, acc[t], false, false);
            acc[t] = __builtin_amdgcn_wmma_f32_16x16x32_f16(
                false, aP[1].h, false, b1.h, (short)0, c0, false, false);
        }
        __syncthreads();
    }

    // ---- Normalize and write out (fp32) ----
#pragma unroll
    for (int r = 0; r < 8; ++r) {
        const float inv  = 1.0f / lrow[r];
        const int   qrow = qbase + wave * 16 + r + 8 * half;
        float* orow = Ob + (size_t)qrow * D_;
#pragma unroll
        for (int t = 0; t < 4; ++t)
            orow[t * 16 + n16] = acc[t][r] * inv;
    }
}

extern "C" void kernel_launch(void* const* d_in, const int* in_sizes, int n_in,
                              void* d_out, int out_size, void* d_ws, size_t ws_size,
                              hipStream_t stream) {
    const float* Q = (const float*)d_in[0];
    const float* K = (const float*)d_in[1];
    const float* V = (const float*)d_in[2];
    const unsigned char* Mask = (const unsigned char*)d_in[3];
    float* out = (float*)d_out;
    (void)in_sizes; (void)n_in; (void)out_size; (void)d_ws; (void)ws_size;

    dim3 grid(S_ / 64, B_ * H_);   // 32 q-tiles x 64 (b,h) heads
    dim3 block(128);               // 4 waves (wave32), one 16-row Q tile each
    hipLaunchKernelGGL(flash_attn_f16wmma, grid, block, 0, stream, Q, K, V, Mask, out);
}
